// DetectionLayer_962072674902
// MI455X (gfx1250) — compile-verified, matrix operand order checked by reading, not run on hardware
//
#include <hip/hip_runtime.h>
#include <cstdint>
#include <cstddef>

#define BATCH 16
#define N_ROIS 1000
#define NUM_CLASSES 81
#define DET_MAX 100
#define MIN_CONF 0.7f
#define NMS_THRESH 0.3f
#define SORT_N 1024

typedef __attribute__((ext_vector_type(16))) _Float16 v16h;
typedef __attribute__((ext_vector_type(8)))  float    v8f;
typedef __attribute__((ext_vector_type(4)))  unsigned v4u;
typedef __attribute__((ext_vector_type(8)))  int      v8i;
typedef __attribute__((ext_vector_type(4)))  int      v4i;

// ---------------------------------------------------------------------------
// Kernel A: per-ROI class argmax + box refinement. One wave32 per ROI.
// ---------------------------------------------------------------------------
__global__ __launch_bounds__(256) void refine_kernel(
    const float* __restrict__ rois,     // [B,N,4]
    const float* __restrict__ probs,    // [B,N,81]
    const float* __restrict__ deltas,   // [B,N,81,4]
    const float* __restrict__ window,   // [B,4]
    float* __restrict__ ws_box,         // [B,N,4]
    float* __restrict__ ws_score,       // [B,N]  (-1 if filtered)
    int*   __restrict__ ws_cls)         // [B,N]
{
  const int wave = (int)((blockIdx.x * blockDim.x + threadIdx.x) >> 5);
  const int lane = (int)(threadIdx.x & 31);
  if (wave >= BATCH * N_ROIS) return;

  const float* p = probs + (size_t)wave * NUM_CLASSES;
  float best = -1.0f;
  int bestc = NUM_CLASSES;
  for (int c = lane; c < NUM_CLASSES; c += 32) {
    float v = p[c];
    if (v > best || (v == best && c < bestc)) { best = v; bestc = c; }
  }
  // wave32 reduction, argmax with lowest-index tiebreak (matches jnp.argmax)
  for (int m = 16; m >= 1; m >>= 1) {
    float ob = __shfl_xor(best, m, 32);
    int   oc = __shfl_xor(bestc, m, 32);
    if (ob > best || (ob == best && oc < bestc)) { best = ob; bestc = oc; }
  }

  if (lane == 0) {
    const int b = wave / N_ROIS;
    const float* roi = rois + (size_t)wave * 4;
    const float* d   = deltas + ((size_t)wave * NUM_CLASSES + bestc) * 4;
    const float* win = window + (size_t)b * 4;

    float y1 = roi[0], x1 = roi[1], y2 = roi[2], x2 = roi[3];
    float h = y2 - y1, w = x2 - x1;
    float cy = y1 + 0.5f * h + d[0] * 0.1f * h;
    float cx = x1 + 0.5f * w + d[1] * 0.1f * w;
    h = h * expf(d[2] * 0.2f);
    w = w * expf(d[3] * 0.2f);
    float ny1 = cy - 0.5f * h, nx1 = cx - 0.5f * w;
    float ny2 = ny1 + h,       nx2 = nx1 + w;
    // clip to window (wy1,wx1,wy2,wx2)
    ny1 = fminf(fmaxf(ny1, win[0]), win[2]);
    nx1 = fminf(fmaxf(nx1, win[1]), win[3]);
    ny2 = fminf(fmaxf(ny2, win[0]), win[2]);
    nx2 = fminf(fmaxf(nx2, win[1]), win[3]);

    float* ob = ws_box + (size_t)wave * 4;
    ob[0] = ny1; ob[1] = nx1; ob[2] = ny2; ob[3] = nx2;
    const bool keep = (bestc > 0) && (best >= MIN_CONF);
    ws_score[wave] = keep ? best : -1.0f;
    ws_cls[wave]   = bestc;
  }
}

// ---------------------------------------------------------------------------
// Kernel B: per-image NMS + top-100. One 1024-thread block per image.
// Boxes staged into LDS with the Tensor Data Mover (tensor_load_to_lds).
// ---------------------------------------------------------------------------
__global__ __launch_bounds__(SORT_N) void nms_kernel(
    const float* __restrict__ ws_box,
    const float* __restrict__ ws_score,
    const int*   __restrict__ ws_cls,
    float* __restrict__ out)            // [B, DET_MAX, 6]
{
  __shared__ float    s_box[N_ROIS * 4];     // 16 KB, TDM destination
  __shared__ float    s_score[SORT_N];
  __shared__ unsigned s_tag[SORT_N];         // slot | (class<<16)
  __shared__ unsigned s_active[SORT_N];
  __shared__ unsigned s_kept[SORT_N];
  __shared__ unsigned s_scan[SORT_N];
  __shared__ int      s_cnt[NUM_CLASSES];
  __shared__ int      s_total;

  const int tid = (int)threadIdx.x;
  const int b   = (int)blockIdx.x;

  // ---- TDM: DMA the 1000x4 f32 box tile Global -> LDS (wave 0 issues) ----
  if (tid < 32) {
    const uint64_t ga = (uint64_t)(uintptr_t)(ws_box + (size_t)b * N_ROIS * 4);
    const unsigned ldsa = (unsigned)(uintptr_t)(void*)s_box;  // LDS byte offset
    const unsigned td0 = N_ROIS * 4;   // 4000 elements, data_size = 4B
    v4u g0;
    g0.x = 1u;                                    // count=1, user descriptor
    g0.y = ldsa;                                  // lds_addr
    g0.z = (unsigned)(ga & 0xFFFFFFFFu);          // global_addr[31:0]
    g0.w = (unsigned)((ga >> 32) & 0x01FFFFFFu) | (2u << 30);  // addr[56:32] | type=2
    v8i g1;
    g1[0] = (int)(2u << 16);                      // data_size=2 (4 bytes), mask=0
    g1[1] = (int)(td0 << 16);                     // tensor_dim0[15:0]
    g1[2] = (int)((td0 >> 16) | (1u << 16));      // tensor_dim0[31:16] | tensor_dim1=1
    g1[3] = (int)(td0 << 16);                     // tile_dim0
    g1[4] = 1;                                    // tile_dim1=1, tile_dim2=0
    g1[5] = (int)td0;                             // tensor_dim0_stride[31:0]
    g1[6] = 0;
    g1[7] = 0;
    v4i gz = {0, 0, 0, 0};
#if defined(__clang_major__) && (__clang_major__ >= 23)
    v8i gz8 = {0, 0, 0, 0, 0, 0, 0, 0};
    __builtin_amdgcn_tensor_load_to_lds(g0, g1, gz, gz, gz8, 0);
#else
    __builtin_amdgcn_tensor_load_to_lds(g0, g1, gz, gz, 0);
#endif
  }

  // ---- stage scores / class tags ----
  if (tid < N_ROIS) {
    s_score[tid] = ws_score[(size_t)b * N_ROIS + tid];
    s_tag[tid]   = (unsigned)tid | ((unsigned)ws_cls[(size_t)b * N_ROIS + tid] << 16);
  } else {
    s_score[tid] = -1.0f;
    s_tag[tid]   = (unsigned)tid;
  }

  if (tid < 32) __builtin_amdgcn_s_wait_tensorcnt(0);
  __syncthreads();

  // ---- bitonic sort (descending by score), payload = tag ----
  for (unsigned k = 2; k <= SORT_N; k <<= 1) {
    for (unsigned j = k >> 1; j > 0; j >>= 1) {
      unsigned ixj = (unsigned)tid ^ j;
      if (ixj > (unsigned)tid) {
        const bool descending = ((tid & k) == 0);
        float a = s_score[tid], c2 = s_score[ixj];
        if ((a < c2) == descending) {
          s_score[tid] = c2; s_score[ixj] = a;
          unsigned t = s_tag[tid]; s_tag[tid] = s_tag[ixj]; s_tag[ixj] = t;
        }
      }
      __syncthreads();
    }
  }

  // ---- init NMS state (positional, must follow sort) ----
  s_active[tid] = (s_score[tid] >= 0.0f) ? 1u : 0u;
  s_kept[tid]   = 0u;
  if (tid < NUM_CLASSES) s_cnt[tid] = 0;
  __syncthreads();

  // ---- greedy per-class NMS over sorted candidates ----
  for (int i = 0; i < SORT_N; ++i) {
    if (s_score[i] < 0.0f) break;       // sorted: rest are sentinels
    if (!s_active[i]) continue;          // suppressed earlier (no writes -> no barrier)
    const unsigned tagi = s_tag[i];
    const int ci     = (int)(tagi >> 16);
    const int slot_i = (int)(tagi & 0xFFFFu);
    const bool kept_i = (s_cnt[ci] < DET_MAX);   // uniform read before write
    __syncthreads();
    if (tid == 0 && kept_i) { s_cnt[ci]++; s_kept[i] = 1u; }
    if (kept_i && tid > i && s_active[tid]) {
      const unsigned tagj = s_tag[tid];
      if ((int)(tagj >> 16) == ci) {
        const float* A  = &s_box[slot_i * 4];
        const float* Bx = &s_box[(int)(tagj & 0xFFFFu) * 4];
        float yy1 = fmaxf(A[0], Bx[0]);
        float xx1 = fmaxf(A[1], Bx[1]);
        float yy2 = fminf(A[2], Bx[2]);
        float xx2 = fminf(A[3], Bx[3]);
        float inter = fmaxf(yy2 - yy1, 0.0f) * fmaxf(xx2 - xx1, 0.0f);
        float areaA = (A[2] - A[0]) * (A[3] - A[1]);
        float areaB = (Bx[2] - Bx[0]) * (Bx[3] - Bx[1]);
        float uni = fmaxf(areaA + areaB - inter, 1e-10f);
        if (inter / uni > NMS_THRESH) s_active[tid] = 0u;
      }
    }
    __syncthreads();
  }
  __syncthreads();

  // ---- total kept count via chained WMMA (wave 0): D = ones(16x32) x flags ----
  // A = splat 1.0 (layout-invariant); each WMMA consumes 512 flags; summing D
  // row 0 over lanes 0..15 gives the permutation-invariant total of all flags.
  if (tid < 32) {
    v16h a, f;
#pragma unroll
    for (int q = 0; q < 16; ++q) a[q] = (_Float16)1.0f;
#pragma unroll
    for (int q = 0; q < 16; ++q) f[q] = (_Float16)(float)s_kept[tid * 16 + q];
    v8f acc = {};
    acc = __builtin_amdgcn_wmma_f32_16x16x32_f16(false, a, false, f,
                                                 (short)0, acc, false, false);
#pragma unroll
    for (int q = 0; q < 16; ++q) f[q] = (_Float16)(float)s_kept[512 + tid * 16 + q];
    acc = __builtin_amdgcn_wmma_f32_16x16x32_f16(false, a, false, f,
                                                 (short)0, acc, false, false);
    float v = acc[0];                       // D row-0 column sums (lanes 0..15)
    v += __shfl_xor(v, 1, 32);
    v += __shfl_xor(v, 2, 32);
    v += __shfl_xor(v, 4, 32);
    v += __shfl_xor(v, 8, 32);
    if (tid == 0) s_total = min(DET_MAX, (int)(v + 0.5f));
  }

  // ---- inclusive prefix scan of kept flags (compaction ranks) ----
  s_scan[tid] = s_kept[tid];
  __syncthreads();
  for (int off = 1; off < SORT_N; off <<= 1) {
    unsigned v = (tid >= off) ? s_scan[tid - off] : 0u;
    __syncthreads();
    s_scan[tid] += v;
    __syncthreads();
  }

  // ---- emit top-DET_MAX detections: [y1,x1,y2,x2,class,score] ----
  if (s_kept[tid]) {
    const unsigned r = s_scan[tid] - 1u;    // exclusive rank (sorted by score)
    if (r < DET_MAX) {
      const unsigned tag = s_tag[tid];
      const int slot = (int)(tag & 0xFFFFu);
      float* o = out + ((size_t)b * DET_MAX + r) * 6;
      o[0] = s_box[slot * 4 + 0];
      o[1] = s_box[slot * 4 + 1];
      o[2] = s_box[slot * 4 + 2];
      o[3] = s_box[slot * 4 + 3];
      o[4] = (float)(tag >> 16);
      o[5] = s_score[tid];
    }
  }
  if (tid < DET_MAX && tid >= s_total) {
    float* o = out + ((size_t)b * DET_MAX + tid) * 6;
#pragma unroll
    for (int q = 0; q < 6; ++q) o[q] = 0.0f;
  }
}

// ---------------------------------------------------------------------------
extern "C" void kernel_launch(void* const* d_in, const int* in_sizes, int n_in,
                              void* d_out, int out_size, void* d_ws, size_t ws_size,
                              hipStream_t stream) {
  (void)in_sizes; (void)n_in; (void)out_size; (void)ws_size;
  const float* rois      = (const float*)d_in[0];
  const float* fpn_class = (const float*)d_in[1];
  const float* fpn_bbox  = (const float*)d_in[2];
  const float* window    = (const float*)d_in[3];
  float* out = (float*)d_out;

  char* ws = (char*)d_ws;
  float* ws_box   = (float*)(ws);                       // B*N*4 f32 = 256000 B
  float* ws_score = (float*)(ws + 256000);              // B*N   f32 =  64000 B
  int*   ws_cls   = (int*)  (ws + 320000);              // B*N   i32 =  64000 B

  // Kernel A: one wave32 per ROI -> 16000 waves, 256 threads/block
  const int total_waves = BATCH * N_ROIS;
  const int blocksA = (total_waves * 32) / 256;         // 2000
  refine_kernel<<<blocksA, 256, 0, stream>>>(rois, fpn_class, fpn_bbox, window,
                                             ws_box, ws_score, ws_cls);

  // Kernel B: one 1024-thread block per image
  nms_kernel<<<BATCH, SORT_N, 0, stream>>>(ws_box, ws_score, ws_cls, out);
}
